// GraphNetworkLtp_21655225106540
// MI455X (gfx1250) — matrix operand order repaired and checked by reading, not compiled, not run on hardware
//
#include <hip/hip_runtime.h>
#include <hip/hip_bf16.h>

#define N_NODES 50000
#define N_EDGES 800000
#define N_GRAPHS 8
#define PITCH 72   // bf16 elements per LDS weight row (144B, 16B aligned, conflict-reduced)
#define LTP 68     // f32 elements per LDS output-tile row (272B: 16B-aligned, bank-stride 4)

typedef __attribute__((ext_vector_type(16))) __bf16 v16bf;
typedef __attribute__((ext_vector_type(8)))  __bf16 v8bf;
typedef __attribute__((ext_vector_type(8)))  float  v8f;

// Hardware f32 -> bf16 convert (RNE); clang lowers pairs to v_cvt_pk_bf16_f32.
__device__ __forceinline__ __bf16 f2bf(float f) { return (__bf16)f; }

// Build a 16x32 bf16 A-fragment (per-lane slice) from 32B-aligned f32 memory.
// p already includes this lane's koff; slots 0..7 <- p[0..7], slots 8..15 <- p[16..23].
__device__ __forceinline__ v16bf load_a_f32(const float* p, float s) {
    const float4 f0 = *(const float4*)(p + 0);
    const float4 f1 = *(const float4*)(p + 4);
    const float4 f2 = *(const float4*)(p + 16);
    const float4 f3 = *(const float4*)(p + 20);
    v16bf a;
    a[0]  = f2bf(f0.x * s); a[1]  = f2bf(f0.y * s); a[2]  = f2bf(f0.z * s); a[3]  = f2bf(f0.w * s);
    a[4]  = f2bf(f1.x * s); a[5]  = f2bf(f1.y * s); a[6]  = f2bf(f1.z * s); a[7]  = f2bf(f1.w * s);
    a[8]  = f2bf(f2.x * s); a[9]  = f2bf(f2.y * s); a[10] = f2bf(f2.z * s); a[11] = f2bf(f2.w * s);
    a[12] = f2bf(f3.x * s); a[13] = f2bf(f3.y * s); a[14] = f2bf(f3.z * s); a[15] = f2bf(f3.w * s);
    return a;
}

// Fused 2-layer MLP + LayerNorm on one 16-row tile, per wave.
//  layer1: K = NS1*32 (A gathered from sp[] pointers, scaled by scl[]), N = 64, +bias, ReLU
//  layer2: K = 64, N = 64, +bias, LayerNorm(gamma=lg, beta=lbe)
// Result: lt[16][LTP] f32 (post-LN). lh is a per-wave 16*PITCH bf16 staging tile.
template <int NS1>
__device__ __forceinline__ void run_mlp(const float* const* sp, const float* scl,
                                        const __bf16* lw1, const __bf16* lw2,
                                        const float* lb1, const float* lb2,
                                        const float* lg,  const float* lbe,
                                        __bf16* lh, float* lt, int lane) {
    const int m = lane & 15, half = lane >> 4, koff = half * 8;
    v8f z = {0.f,0.f,0.f,0.f,0.f,0.f,0.f,0.f};
    v8f acc[4]; acc[0] = z; acc[1] = z; acc[2] = z; acc[3] = z;

    #pragma unroll
    for (int s = 0; s < NS1; ++s) {
        v16bf a = load_a_f32(sp[s] + koff, scl[s]);
        const __bf16* wr = lw1 + (s * 32 + koff) * PITCH + m;
        #pragma unroll
        for (int nt = 0; nt < 4; ++nt) {
            v16bf b;
            #pragma unroll
            for (int j = 0; j < 16; ++j) b[j] = wr[((j & 7) + (j >> 3) * 16) * PITCH + nt * 16];
            acc[nt] = __builtin_amdgcn_wmma_f32_16x16x32_bf16(
                false, a, false, b, (short)0, acc[nt], false, false);
        }
    }
    // bias + ReLU, re-layout C/D -> A via LDS (bf16)
    #pragma unroll
    for (int nt = 0; nt < 4; ++nt) {
        const int n = nt * 16 + m;
        const float bv = lb1[n];
        #pragma unroll
        for (int gi = 0; gi < 8; ++gi) {
            float v = acc[nt][gi] + bv;
            v = v > 0.f ? v : 0.f;
            lh[(half * 8 + gi) * PITCH + n] = f2bf(v);
        }
    }
    asm volatile("s_wait_dscnt 0" ::: "memory");

    acc[0] = z; acc[1] = z; acc[2] = z; acc[3] = z;
    #pragma unroll
    for (int s = 0; s < 2; ++s) {
        const __bf16* hp = lh + m * PITCH + s * 32 + koff;
        v8bf lo = *(const v8bf*)(hp);
        v8bf hi = *(const v8bf*)(hp + 16);
        v16bf a = __builtin_shufflevector(lo, hi, 0,1,2,3,4,5,6,7,8,9,10,11,12,13,14,15);
        const __bf16* wr = lw2 + (s * 32 + koff) * PITCH + m;
        #pragma unroll
        for (int nt = 0; nt < 4; ++nt) {
            v16bf b;
            #pragma unroll
            for (int j = 0; j < 16; ++j) b[j] = wr[((j & 7) + (j >> 3) * 16) * PITCH + nt * 16];
            acc[nt] = __builtin_amdgcn_wmma_f32_16x16x32_bf16(
                false, a, false, b, (short)0, acc[nt], false, false);
        }
    }
    #pragma unroll
    for (int nt = 0; nt < 4; ++nt) {
        const int n = nt * 16 + m;
        const float bv = lb2[n];
        #pragma unroll
        for (int gi = 0; gi < 8; ++gi)
            lt[(half * 8 + gi) * LTP + n] = acc[nt][gi] + bv;
    }
    asm volatile("s_wait_dscnt 0" ::: "memory");

    // LayerNorm: lanes 0..15 each own one row of 64 (float4 / ds_load_b128 path)
    if (half == 0) {
        float4* rp4 = (float4*)(lt + m * LTP);
        float4 s4 = rp4[0];
        #pragma unroll
        for (int q = 1; q < 16; ++q) {
            float4 v = rp4[q];
            s4.x += v.x; s4.y += v.y; s4.z += v.z; s4.w += v.w;
        }
        const float mean = (s4.x + s4.y + s4.z + s4.w) * (1.f / 64.f);
        float var = 0.f;
        #pragma unroll
        for (int q = 0; q < 16; ++q) {
            float4 v = rp4[q];
            float dx = v.x - mean, dy = v.y - mean, dz = v.z - mean, dw = v.w - mean;
            var += dx * dx + dy * dy + dz * dz + dw * dw;
        }
        var *= (1.f / 64.f);
        const float inv = rsqrtf(var + 1e-5f);
        const float4* g4 = (const float4*)lg;
        const float4* b4 = (const float4*)lbe;
        #pragma unroll
        for (int q = 0; q < 16; ++q) {
            float4 v = rp4[q], gv = g4[q], bv = b4[q];
            v.x = (v.x - mean) * inv * gv.x + bv.x;
            v.y = (v.y - mean) * inv * gv.y + bv.y;
            v.z = (v.z - mean) * inv * gv.z + bv.z;
            v.w = (v.w - mean) * inv * gv.w + bv.w;
            rp4[q] = v;
        }
    }
    asm volatile("s_wait_dscnt 0" ::: "memory");
}

// -------------------- degree count --------------------
__global__ void count_kernel(const int* __restrict__ row, float* __restrict__ cnt) {
    int e = blockIdx.x * blockDim.x + threadIdx.x;
    if (e < N_EDGES) atomicAdd(&cnt[row[e]], 1.0f);
}

// -------------------- edge MLP: [x[row],x[col],ea,u[eb]] (256) -> 64, + edges_agg ----
__global__ __launch_bounds__(64) void edge_kernel(
    const float* __restrict__ x, const float* __restrict__ ea_in, const float* __restrict__ u,
    const int* __restrict__ row, const int* __restrict__ col, const int* __restrict__ batch,
    const float* __restrict__ w1, const float* __restrict__ b1,
    const float* __restrict__ w2, const float* __restrict__ b2,
    const float* __restrict__ g, const float* __restrict__ be,
    float* __restrict__ ea_out, float* __restrict__ edges_agg) {
    __shared__ __align__(16) __bf16 lw1[256 * PITCH];
    __shared__ __align__(16) __bf16 lw2[64 * PITCH];
    __shared__ __align__(16) float lb1[64], lb2[64], lg[64], lbe[64];
    __shared__ __align__(16) __bf16 lh[2][16 * PITCH];
    __shared__ __align__(16) float lt[2][16 * LTP];
    __shared__ int   leb[2][16];
    __shared__ float leagg[N_GRAPHS * 64];

    const int tid = threadIdx.x, lane = tid & 31, wave = tid >> 5;
    const int m = lane & 15, half = lane >> 4;
    for (int i = tid; i < 256 * 64; i += 64) { int k = i >> 6, n = i & 63; lw1[k * PITCH + n] = f2bf(w1[i]); }
    for (int i = tid; i < 64 * 64;  i += 64) { int k = i >> 6, n = i & 63; lw2[k * PITCH + n] = f2bf(w2[i]); }
    if (tid < 64) { lb1[tid] = b1[tid]; lb2[tid] = b2[tid]; lg[tid] = g[tid]; lbe[tid] = be[tid]; }
    for (int i = tid; i < N_GRAPHS * 64; i += 64) leagg[i] = 0.f;
    __syncthreads();

    const int gw = (blockIdx.x * 64 + tid) >> 5;
    const int nw = gridDim.x * 2;
    float scl[8] = {1.f,1.f,1.f,1.f,1.f,1.f,1.f,1.f};
    for (int tile = gw; tile < N_EDGES / 16; tile += nw) {
        const int e  = tile * 16 + m;
        const int rm = row[e], cm = col[e];
        const int gb = batch[rm];
        if (half == 0) leb[wave][m] = gb;
        const float* px = x + (size_t)rm * 64;
        const float* pc = x + (size_t)cm * 64;
        const float* pa = ea_in + (size_t)e * 64;
        const float* pu = u + (size_t)gb * 64;
        const float* sp[8] = {px, px + 32, pc, pc + 32, pa, pa + 32, pu, pu + 32};
        run_mlp<8>(sp, scl, lw1, lw2, lb1, lb2, lg, lbe, lh[wave], lt[wave], lane);

        float4* outp4 = (float4*)(ea_out + (size_t)tile * 1024);
        #pragma unroll
        for (int t4 = lane; t4 < 256; t4 += 32) {            // 256 float4 = 16x64 tile
            const int mr = t4 >> 4, nq = (t4 & 15) * 4;
            const float4 v = *(const float4*)&lt[wave][mr * LTP + nq];
            outp4[t4] = v;                                   // coalesced b128 store
            float* ag = &leagg[leb[wave][mr] * 64 + nq];
            atomicAdd(ag + 0, v.x); atomicAdd(ag + 1, v.y);
            atomicAdd(ag + 2, v.z); atomicAdd(ag + 3, v.w);
        }
    }
    __syncthreads();
    for (int i = tid; i < N_GRAPHS * 64; i += 64) atomicAdd(&edges_agg[i], leagg[i]);
}

// -------------------- msg MLP: [x[col],ea_new] (128) -> 64, scatter-add into agg ----
__global__ __launch_bounds__(64) void msg_kernel(
    const float* __restrict__ x, const float* __restrict__ ea,
    const int* __restrict__ row, const int* __restrict__ col,
    const float* __restrict__ w1, const float* __restrict__ b1,
    const float* __restrict__ w2, const float* __restrict__ b2,
    const float* __restrict__ g, const float* __restrict__ be,
    float* __restrict__ agg) {
    __shared__ __align__(16) __bf16 lw1[128 * PITCH];
    __shared__ __align__(16) __bf16 lw2[64 * PITCH];
    __shared__ __align__(16) float lb1[64], lb2[64], lg[64], lbe[64];
    __shared__ __align__(16) __bf16 lh[2][16 * PITCH];
    __shared__ __align__(16) float lt[2][16 * LTP];
    __shared__ int   lrow[2][16];

    const int tid = threadIdx.x, lane = tid & 31, wave = tid >> 5;
    const int m = lane & 15, half = lane >> 4;
    for (int i = tid; i < 128 * 64; i += 64) { int k = i >> 6, n = i & 63; lw1[k * PITCH + n] = f2bf(w1[i]); }
    for (int i = tid; i < 64 * 64;  i += 64) { int k = i >> 6, n = i & 63; lw2[k * PITCH + n] = f2bf(w2[i]); }
    if (tid < 64) { lb1[tid] = b1[tid]; lb2[tid] = b2[tid]; lg[tid] = g[tid]; lbe[tid] = be[tid]; }
    __syncthreads();

    const int gw = (blockIdx.x * 64 + tid) >> 5;
    const int nw = gridDim.x * 2;
    float scl[4] = {1.f,1.f,1.f,1.f};
    for (int tile = gw; tile < N_EDGES / 16; tile += nw) {
        const int e  = tile * 16 + m;
        const int rm = row[e], cm = col[e];
        if (half == 0) lrow[wave][m] = rm;
        const float* pc = x + (size_t)cm * 64;
        const float* pa = ea + (size_t)e * 64;
        const float* sp[4] = {pc, pc + 32, pa, pa + 32};
        run_mlp<4>(sp, scl, lw1, lw2, lb1, lb2, lg, lbe, lh[wave], lt[wave], lane);

        #pragma unroll
        for (int t4 = lane; t4 < 256; t4 += 32) {
            const int mr = t4 >> 4, nq = (t4 & 15) * 4;
            const float4 v = *(const float4*)&lt[wave][mr * LTP + nq];
            float* dst = &agg[(size_t)lrow[wave][mr] * 64 + nq];
            atomicAdd(dst + 0, v.x); atomicAdd(dst + 1, v.y);
            atomicAdd(dst + 2, v.z); atomicAdd(dst + 3, v.w);
        }
    }
}

// -------------------- node MLP: [x, agg/cnt, u[batch]] (192) -> 64, + nodes_agg ----
__global__ __launch_bounds__(64) void node_kernel(
    const float* __restrict__ x, const float* __restrict__ agg,
    const float* __restrict__ cnt, const float* __restrict__ u,
    const int* __restrict__ batch,
    const float* __restrict__ w1, const float* __restrict__ b1,
    const float* __restrict__ w2, const float* __restrict__ b2,
    const float* __restrict__ g, const float* __restrict__ be,
    float* __restrict__ x_out, float* __restrict__ nodes_agg) {
    __shared__ __align__(16) __bf16 lw1[192 * PITCH];
    __shared__ __align__(16) __bf16 lw2[64 * PITCH];
    __shared__ __align__(16) float lb1[64], lb2[64], lg[64], lbe[64];
    __shared__ __align__(16) __bf16 lh[2][16 * PITCH];
    __shared__ __align__(16) float lt[2][16 * LTP];
    __shared__ int   lbt[2][16];
    __shared__ float lnagg[N_GRAPHS * 64];

    const int tid = threadIdx.x, lane = tid & 31, wave = tid >> 5;
    const int m = lane & 15, half = lane >> 4;
    for (int i = tid; i < 192 * 64; i += 64) { int k = i >> 6, n = i & 63; lw1[k * PITCH + n] = f2bf(w1[i]); }
    for (int i = tid; i < 64 * 64;  i += 64) { int k = i >> 6, n = i & 63; lw2[k * PITCH + n] = f2bf(w2[i]); }
    if (tid < 64) { lb1[tid] = b1[tid]; lb2[tid] = b2[tid]; lg[tid] = g[tid]; lbe[tid] = be[tid]; }
    for (int i = tid; i < N_GRAPHS * 64; i += 64) lnagg[i] = 0.f;
    __syncthreads();

    const int gw = (blockIdx.x * 64 + tid) >> 5;
    const int nw = gridDim.x * 2;
    for (int tile = gw; tile < N_NODES / 16; tile += nw) {
        const int i  = tile * 16 + m;
        const int gb = batch[i];
        if (half == 0) lbt[wave][m] = gb;
        const float inv = 1.f / fmaxf(cnt[i], 1.f);
        const float* p0 = x   + (size_t)i * 64;
        const float* p1 = agg + (size_t)i * 64;
        const float* p2 = u + (size_t)gb * 64;
        const float* sp[6] = {p0, p0 + 32, p1, p1 + 32, p2, p2 + 32};
        float scl[6] = {1.f, 1.f, inv, inv, 1.f, 1.f};
        run_mlp<6>(sp, scl, lw1, lw2, lb1, lb2, lg, lbe, lh[wave], lt[wave], lane);

        float4* outp4 = (float4*)(x_out + (size_t)tile * 1024);
        #pragma unroll
        for (int t4 = lane; t4 < 256; t4 += 32) {
            const int mr = t4 >> 4, nq = (t4 & 15) * 4;
            const float4 v = *(const float4*)&lt[wave][mr * LTP + nq];
            outp4[t4] = v;
            float* ag = &lnagg[lbt[wave][mr] * 64 + nq];
            atomicAdd(ag + 0, v.x); atomicAdd(ag + 1, v.y);
            atomicAdd(ag + 2, v.z); atomicAdd(ag + 3, v.w);
        }
    }
    __syncthreads();
    for (int i = tid; i < N_GRAPHS * 64; i += 64) atomicAdd(&nodes_agg[i], lnagg[i]);
}

// -------------------- global MLPs (single block): u1 = MLP(u); u' = MLP([u1,na,ea]) ----
__global__ __launch_bounds__(64) void glob_kernel(
    const float* __restrict__ u, const float* __restrict__ nodes_agg,
    const float* __restrict__ edges_agg,
    const float* __restrict__ g1w1, const float* __restrict__ g1b1,
    const float* __restrict__ g1w2, const float* __restrict__ g1b2,
    const float* __restrict__ g1g,  const float* __restrict__ g1be,
    const float* __restrict__ g2w1, const float* __restrict__ g2b1,
    const float* __restrict__ g2w2, const float* __restrict__ g2b2,
    const float* __restrict__ g2g,  const float* __restrict__ g2be,
    float* __restrict__ u_out) {
    __shared__ __align__(16) __bf16 lw[256 * PITCH];
    __shared__ __align__(16) float lb1[64], lb2[64], lg[64], lbe[64];
    __shared__ __align__(16) __bf16 lh[16 * PITCH];
    __shared__ __align__(16) float lt[16 * LTP];
    __shared__ __align__(16) float lu1[N_GRAPHS * 64];
    __shared__ __align__(16) float zrow[64];

    const int tid = threadIdx.x, lane = tid & 31, wave = tid >> 5;
    // phase A: glob1 (64 -> 64 -> 64) on u (8 rows, padded to 16 with zeros)
    for (int i = tid; i < 64 * 64; i += 64) { int k = i >> 6, n = i & 63; lw[k * PITCH + n] = f2bf(g1w1[i]); }
    for (int i = tid; i < 64 * 64; i += 64) { int k = i >> 6, n = i & 63; lw[(64 + k) * PITCH + n] = f2bf(g1w2[i]); }
    if (tid < 64) { lb1[tid] = g1b1[tid]; lb2[tid] = g1b2[tid]; lg[tid] = g1g[tid]; lbe[tid] = g1be[tid]; zrow[tid] = 0.f; }
    __syncthreads();
    if (wave == 0) {
        const int m = lane & 15;
        const float* base = (m < N_GRAPHS) ? (u + (size_t)m * 64) : zrow;
        const float* sp[2] = {base, base + 32};
        float scl[2] = {1.f, 1.f};
        run_mlp<2>(sp, scl, lw, lw + 64 * PITCH, lb1, lb2, lg, lbe, lh, lt, lane);
        for (int t = lane; t < N_GRAPHS * 64; t += 32) lu1[t] = lt[(t >> 6) * LTP + (t & 63)];
    }
    __syncthreads();
    // phase B: glob2 ([u1, nodes_agg, edges_agg] 192 -> 64 -> 64)
    for (int i = tid; i < 192 * 64; i += 64) { int k = i >> 6, n = i & 63; lw[k * PITCH + n] = f2bf(g2w1[i]); }
    for (int i = tid; i < 64 * 64;  i += 64) { int k = i >> 6, n = i & 63; lw[(192 + k) * PITCH + n] = f2bf(g2w2[i]); }
    if (tid < 64) { lb1[tid] = g2b1[tid]; lb2[tid] = g2b2[tid]; lg[tid] = g2g[tid]; lbe[tid] = g2be[tid]; }
    __syncthreads();
    if (wave == 0) {
        const int m = lane & 15;
        const float* q0 = (m < N_GRAPHS) ? (lu1 + (size_t)m * 64) : zrow;
        const float* q1 = (m < N_GRAPHS) ? (nodes_agg + (size_t)m * 64) : zrow;
        const float* q2 = (m < N_GRAPHS) ? (edges_agg + (size_t)m * 64) : zrow;
        const float* sp[6] = {q0, q0 + 32, q1, q1 + 32, q2, q2 + 32};
        float scl[6] = {1.f, 1.f, 1.f, 1.f, 1.f, 1.f};
        run_mlp<6>(sp, scl, lw, lw + 192 * PITCH, lb1, lb2, lg, lbe, lh, lt, lane);
        for (int t = lane; t < N_GRAPHS * 64; t += 32) u_out[t] = lt[(t >> 6) * LTP + (t & 63)];
    }
}

extern "C" void kernel_launch(void* const* d_in, const int* in_sizes, int n_in,
                              void* d_out, int out_size, void* d_ws, size_t ws_size,
                              hipStream_t stream) {
    // setup_inputs() insertion order; params leaves in nested dict order
    // {edge,node1,node2,glob1,glob2} x {w1,b1,w2,b2,g,beta}
    const float* x_in  = (const float*)d_in[0];
    const float* ea_in = (const float*)d_in[1];
    const float* u_in  = (const float*)d_in[2];
    const float* P[30];
    for (int i = 0; i < 30; ++i) P[i] = (const float*)d_in[3 + i];
    const int* eidx  = (const int*)d_in[33];
    const int* batch = (const int*)d_in[34];
    const int* row = eidx;
    const int* col = eidx + N_EDGES;

    float* x_out  = (float*)d_out;
    float* ea_out = x_out + (size_t)N_NODES * 64;
    float* u_out  = ea_out + (size_t)N_EDGES * 64;

    float* agg       = (float*)d_ws;                       // N_NODES*64
    float* cnt       = agg + (size_t)N_NODES * 64;         // N_NODES
    float* nodes_agg = cnt + N_NODES;                      // 512
    float* edges_agg = nodes_agg + N_GRAPHS * 64;          // 512

    hipMemsetAsync(cnt, 0, N_NODES * sizeof(float), stream);
    count_kernel<<<(N_EDGES + 255) / 256, 256, 0, stream>>>(row, cnt);

    // per-MLP K1 dims for round-slicing of w1
    const int K1[5] = {256, 128, 192, 64, 192};
    for (int r = 0; r < 3; ++r) {
        const float* xc  = r ? x_out  : x_in;
        const float* eac = r ? ea_out : ea_in;
        const float* uc  = r ? u_out  : u_in;

        hipMemsetAsync(agg, 0, (size_t)N_NODES * 64 * sizeof(float), stream);
        hipMemsetAsync(nodes_agg, 0, 2 * N_GRAPHS * 64 * sizeof(float), stream);

        // round-r parameter slices: w1[r], b1[r], w2[r], b2[r], g[r], beta[r]
        const float* pr[5][6];
        for (int mlp = 0; mlp < 5; ++mlp) {
            const float* const* b = P + mlp * 6;
            pr[mlp][0] = b[0] + (size_t)r * K1[mlp] * 64;
            pr[mlp][1] = b[1] + (size_t)r * 64;
            pr[mlp][2] = b[2] + (size_t)r * 64 * 64;
            pr[mlp][3] = b[3] + (size_t)r * 64;
            pr[mlp][4] = b[4] + (size_t)r * 64;
            pr[mlp][5] = b[5] + (size_t)r * 64;
        }

        edge_kernel<<<1024, 64, 0, stream>>>(xc, eac, uc, row, col, batch,
            pr[0][0], pr[0][1], pr[0][2], pr[0][3], pr[0][4], pr[0][5],
            ea_out, edges_agg);
        msg_kernel<<<1024, 64, 0, stream>>>(xc, ea_out, row, col,
            pr[1][0], pr[1][1], pr[1][2], pr[1][3], pr[1][4], pr[1][5],
            agg);
        node_kernel<<<256, 64, 0, stream>>>(xc, agg, cnt, uc, batch,
            pr[2][0], pr[2][1], pr[2][2], pr[2][3], pr[2][4], pr[2][5],
            x_out, nodes_agg);
        glob_kernel<<<1, 64, 0, stream>>>(uc, nodes_agg, edges_agg,
            pr[3][0], pr[3][1], pr[3][2], pr[3][3], pr[3][4], pr[3][5],
            pr[4][0], pr[4][1], pr[4][2], pr[4][3], pr[4][4], pr[4][5],
            u_out);
    }
}
